// GNNModel_2241972928748
// MI455X (gfx1250) — compile-verified, matrix operand order checked by reading, not run on hardware
//
#include <hip/hip_runtime.h>
#include <hip/hip_bf16.h>

typedef __attribute__((ext_vector_type(16))) _Float16 v16h;
typedef __attribute__((ext_vector_type(8)))  _Float16 v8h;
typedef __attribute__((ext_vector_type(8)))  float    v8f;

#define DEV_ATOMIC_ADD_F32(p, v) \
    __hip_atomic_fetch_add((p), (v), __ATOMIC_RELAXED, __HIP_MEMORY_SCOPE_AGENT)

// ---------------------------------------------------------------------------
// Utility kernels
// ---------------------------------------------------------------------------
__global__ void fill_f32_kernel(float* __restrict__ p, float v, int n) {
    int i = blockIdx.x * blockDim.x + threadIdx.x;
    if (i < n) p[i] = v;
}

__global__ void cvt_f16_kernel(const float* __restrict__ x, _Float16* __restrict__ xh, int n) {
    int i = blockIdx.x * blockDim.x + threadIdx.x;
    if (i < n) xh[i] = (_Float16)x[i];
}

__global__ void deg_kernel(const long long* __restrict__ dst, float* __restrict__ deg, int nE) {
    int e = blockIdx.x * blockDim.x + threadIdx.x;
    if (e < nE) DEV_ATOMIC_ADD_F32(&deg[dst[e]], 1.0f);
}

__global__ void rsqrt_kernel(const float* __restrict__ deg, float* __restrict__ dinv, int n) {
    int i = blockIdx.x * blockDim.x + threadIdx.x;
    if (i < n) dinv[i] = rsqrtf(deg[i]);   // deg >= 1 always (self loop)
}

// ---------------------------------------------------------------------------
// WMMA GEMM: C[N, DOUT] = A[N, 128](f16) x W[128, DOUT](f32, converted in LDS)
// One wave computes a 16-row strip across all DOUT columns.
// ---------------------------------------------------------------------------
template <int DOUT>
__global__ __launch_bounds__(256) void gemm_wmma_kernel(
    const _Float16* __restrict__ A, const float* __restrict__ W,
    float* __restrict__ C, int nRows)
{
    __shared__ _Float16 Wt[DOUT * 128];     // transposed: Wt[col*128 + k]
    for (int i = threadIdx.x; i < 128 * DOUT; i += 256) {
        int k = i / DOUT, c = i % DOUT;
        Wt[c * 128 + k] = (_Float16)W[i];
    }
    __syncthreads();

    const int wave = threadIdx.x >> 5;
    const int lane = threadIdx.x & 31;
    const int tile = blockIdx.x * 8 + wave;          // 16-row tile index
    if (tile >= nRows / 16) return;
    const int half = lane >> 4;                      // 0: lanes 0-15, 1: lanes 16-31
    const int l16  = lane & 15;
    const int row  = tile * 16 + l16;

    // A fragments: 16-bit A 16x32 layout. Lane group `half` holds
    // K = k0 + half*8 + 0..7 (halves 0-7) and K = k0 + 16 + half*8 + 0..7 (halves 8-15).
    v16h a[4];
    const _Float16* arow = A + (size_t)row * 128;
#pragma unroll
    for (int kt = 0; kt < 4; ++kt) {
        const v8h lo = *(const v8h*)(arow + kt * 32 + half * 8);
        const v8h hi = *(const v8h*)(arow + kt * 32 + 16 + half * 8);
#pragma unroll
        for (int j = 0; j < 8; ++j) { a[kt][j] = lo[j]; a[kt][j + 8] = hi[j]; }
    }

#pragma unroll
    for (int nt = 0; nt < DOUT / 16; ++nt) {
        const int col = nt * 16 + l16;
        const _Float16* wcol = Wt + col * 128;       // contiguous in K
        v8f c = {};
#pragma unroll
        for (int kt = 0; kt < 4; ++kt) {
            const v8h blo = *(const v8h*)(wcol + kt * 32 + half * 8);
            const v8h bhi = *(const v8h*)(wcol + kt * 32 + 16 + half * 8);
            v16h b;
#pragma unroll
            for (int j = 0; j < 8; ++j) { b[j] = blo[j]; b[j + 8] = bhi[j]; }
            c = __builtin_amdgcn_wmma_f32_16x16x32_f16(
                    false, a[kt], false, b, (short)0, c, false, false);
        }
        // C/D layout: VGPR r, lanes 0-15 -> M=r, lanes 16-31 -> M=r+8, N=l16
        float* crow = C + (size_t)(tile * 16 + half * 8) * DOUT + col;
#pragma unroll
        for (int r = 0; r < 8; ++r) crow[(size_t)r * DOUT] = c[r];
    }
}

// ---------------------------------------------------------------------------
// Edge scatter: one wave per edge, D/32 floats per lane, f32 atomics (L2-resident)
// ---------------------------------------------------------------------------
template <int D>
__global__ __launch_bounds__(256) void scatter_kernel(
    const long long* __restrict__ src, const long long* __restrict__ dst,
    const float* __restrict__ dinv, const float* __restrict__ h,
    float* __restrict__ agg, int nE)
{
    constexpr int VEC = D / 32;
    const int e    = (blockIdx.x * 256 + threadIdx.x) >> 5;
    const int lane = threadIdx.x & 31;
    if (e >= nE) return;
    const long long s = src[e], d = dst[e];
    const float norm = dinv[s] * dinv[d];
    const float* hp = h   + (size_t)s * D + lane * VEC;
    float*       op = agg + (size_t)d * D + lane * VEC;
#pragma unroll
    for (int i = 0; i < VEC; ++i)
        DEV_ATOMIC_ADD_F32(&op[i], hp[i] * norm);
}

// ---------------------------------------------------------------------------
// Finalize layer 1: add self-loop term + bias, ReLU, emit f16 for GEMM2
// ---------------------------------------------------------------------------
__global__ void finalize1_kernel(const float* __restrict__ agg, const float* __restrict__ h,
                                 const float* __restrict__ dinv, const float* __restrict__ b,
                                 _Float16* __restrict__ outh, int total)
{
    int i = blockIdx.x * blockDim.x + threadIdx.x;
    if (i >= total) return;
    int node = i >> 7, col = i & 127;                // D = 128
    float di = dinv[node];
    float v = agg[i] + di * di * h[i] + b[col];
    outh[i] = (_Float16)fmaxf(v, 0.0f);
}

// Finalize layer 2: out += self-loop term + bias (out already holds edge sums)
__global__ void finalize2_kernel(float* __restrict__ out, const float* __restrict__ g,
                                 const float* __restrict__ dinv, const float* __restrict__ b,
                                 int total)
{
    int i = blockIdx.x * blockDim.x + threadIdx.x;
    if (i >= total) return;
    int node = i >> 6, col = i & 63;                 // D = 64
    float di = dinv[node];
    out[i] = out[i] + di * di * g[i] + b[col];
}

// ---------------------------------------------------------------------------
// Host launcher
// ---------------------------------------------------------------------------
extern "C" void kernel_launch(void* const* d_in, const int* in_sizes, int n_in,
                              void* d_out, int out_size, void* d_ws, size_t ws_size,
                              hipStream_t stream)
{
    const float*      x  = (const float*)d_in[0];
    const long long*  ei = (const long long*)d_in[1];     // int64 [2, E]
    const float*      W1 = (const float*)d_in[2];
    const float*      b1 = (const float*)d_in[3];
    const float*      W2 = (const float*)d_in[4];
    const float*      b2 = (const float*)d_in[5];
    float* out = (float*)d_out;

    const int DIN = 128, DHID = 128, DOUT = 64;
    const int N = in_sizes[0] / DIN;
    const int E = in_sizes[1] / 2;
    const long long* srcI = ei;
    const long long* dstI = ei + E;

    // Workspace bump allocation (256B aligned)
    char* base = (char*)d_ws;
    size_t off = 0;
    auto alloc = [&](size_t bytes) { size_t o = off; off = (off + bytes + 255) & ~(size_t)255; return o; };
    float*     deg  = (float*)(base + alloc((size_t)N * 4));
    float*     dinv = (float*)(base + alloc((size_t)N * 4));
    _Float16*  xh   = (_Float16*)(base + alloc((size_t)N * DIN * 2));  // reused as hh after GEMM1
    float*     h1   = (float*)(base + alloc((size_t)N * DHID * 4));
    float*     agg1 = (float*)(base + alloc((size_t)N * DHID * 4));
    float*     g    = (float*)(base + alloc((size_t)N * DOUT * 4));
    _Float16*  hh   = xh;   // xh dead after GEMM1; alias for layer-2 input
    (void)ws_size; (void)n_in; (void)out_size;

    const int B = 256;
    auto cdiv = [](int a, int b) { return (a + b - 1) / b; };

    // 1. Init accumulators (must re-zero every call for graph replay determinism)
    fill_f32_kernel<<<cdiv(N, B), B, 0, stream>>>(deg, 1.0f, N);                       // self loop
    fill_f32_kernel<<<cdiv(N * DHID, B), B, 0, stream>>>(agg1, 0.0f, N * DHID);
    fill_f32_kernel<<<cdiv(N * DOUT, B), B, 0, stream>>>(out, 0.0f, N * DOUT);

    // 2. Convert x to f16
    cvt_f16_kernel<<<cdiv(N * DIN, B), B, 0, stream>>>(x, xh, N * DIN);

    // 3. Degrees and normalization
    deg_kernel<<<cdiv(E, B), B, 0, stream>>>(dstI, deg, E);
    rsqrt_kernel<<<cdiv(N, B), B, 0, stream>>>(deg, dinv, N);

    // 4. GEMM1: h1 = xh @ W1  (v_wmma_f32_16x16x32_f16)
    const int tiles = N / 16;                  // N = 50000 -> 3125 tiles exactly
    gemm_wmma_kernel<DHID><<<cdiv(tiles, 8), B, 0, stream>>>(xh, W1, h1, N);

    // 5. Scatter layer 1: agg1[dst] += norm * h1[src]
    scatter_kernel<DHID><<<cdiv(E * 32, B), B, 0, stream>>>(srcI, dstI, dinv, h1, agg1, E);

    // 6. Finalize layer 1 (self loop + bias + ReLU) -> f16 input for GEMM2
    finalize1_kernel<<<cdiv(N * DHID, B), B, 0, stream>>>(agg1, h1, dinv, b1, hh, N * DHID);

    // 7. GEMM2: g = hh @ W2
    gemm_wmma_kernel<DOUT><<<cdiv(tiles, 8), B, 0, stream>>>(hh, W2, g, N);

    // 8. Scatter layer 2 directly into d_out
    scatter_kernel<DOUT><<<cdiv(E * 32, B), B, 0, stream>>>(srcI, dstI, dinv, g, out, E);

    // 9. Finalize layer 2 (self loop + bias)
    finalize2_kernel<<<cdiv(N * DOUT, B), B, 0, stream>>>(out, g, dinv, b2, N * DOUT);
}
